// DenoisingTransformer_20899310862465
// MI455X (gfx1250) — matrix operand order repaired and checked by reading, not compiled, not run on hardware
//
#include <hip/hip_runtime.h>

typedef _Float16 half_t;
typedef __attribute__((ext_vector_type(4)))  _Float16 v4h;
typedef __attribute__((ext_vector_type(8)))  _Float16 v8h;
typedef __attribute__((ext_vector_type(16))) _Float16 v16h;
typedef __attribute__((ext_vector_type(8)))  float    v8f;
typedef __attribute__((ext_vector_type(4)))  float    v4f;

#define D_MODEL 1024
#define S_LEN   4096
#define BATCH   4
#define NHEADS  16
#define HDIM    64
#define NROWS   (BATCH*S_LEN)            /* 16384 */
#define LOG2_THETA 13.287712379549449f   /* log2(10000) */

constexpr int BM = 128, BN = 128, BK = 32;
constexpr int LDT = 40;   // 32 + 8 halves pad (bank-conflict-free fragment loads)
constexpr int LDVK = 72;  // 64 + 8 halves pad

// ---------------- TDM support (Tensor Data Mover, CDNA5) --------------------
#if defined(__has_builtin)
#if __has_builtin(__builtin_amdgcn_tensor_load_to_lds) && \
    __has_builtin(__builtin_amdgcn_s_wait_tensorcnt)
#define HAVE_TDM 1
#endif
#endif
#ifndef HAVE_TDM
#define HAVE_TDM 0
#endif

#if HAVE_TDM
typedef __attribute__((ext_vector_type(4))) unsigned int u32x4;
typedef __attribute__((ext_vector_type(8))) int          i32x8;
typedef __attribute__((ext_vector_type(4))) int          i32x4;

// Load a 128-row x 32-half tile (row stride 1024 halves) into LDS with
// 4-DWORD padding after every 16 DWORDs -> LDS row stride = 40 halves (LDT).
// D# fields per CDNA5 ISA 8.3/8.4.
__device__ __forceinline__ void tdm_load_tile_128x32(const half_t* g,
                                                     unsigned lds_off) {
  unsigned long long ga = (unsigned long long)(uintptr_t)g;
  u32x4 g0;
  g0[0] = 1u;                                   // count=1 (user descriptor)
  g0[1] = lds_off;                              // lds_addr (bytes)
  g0[2] = (unsigned)ga;                         // global_addr[31:0]
  g0[3] = (unsigned)((ga >> 32) & 0x01FFFFFFu)  // global_addr[56:32]
          | 0x80000000u;                        // type=2 ("image")
  i32x8 g1;
  g1[0] = (1 << 16)      // data_size = 2 bytes
        | (1 << 20)      // pad_enable
        | (3 << 22)      // pad_interval: 16 DWORDs
        | (3 << 25);     // pad_amount: 4 DWORDs
  g1[1] = (int)(1024u << 16);   // tensor_dim0 = 1024 (bits 79:48, low half)
  g1[2] = (int)(16384u << 16);  // tensor_dim1 = 16384 (bits 111:80, low half)
  g1[3] = (int)(32u << 16);     // tile_dim0 = 32 (bits 127:112)
  g1[4] = 128;                  // tile_dim1 = 128; tile_dim2 = 0
  g1[5] = 1024;                 // tensor_dim0_stride = 1024
  g1[6] = 0;
  g1[7] = 0;
  i32x4 z4 = {0, 0, 0, 0};
#if __clang_major__ >= 23
  i32x8 z8 = {0, 0, 0, 0, 0, 0, 0, 0};
  __builtin_amdgcn_tensor_load_to_lds(g0, g1, z4, z4, z8, 0);
#else
  __builtin_amdgcn_tensor_load_to_lds(g0, g1, z4, z4, 0);
#endif
}
__device__ __forceinline__ unsigned lds_addr_of(const void* p) {
  return (unsigned)(uintptr_t)p;   // generic LDS pointer: low 32 bits = offset
}
#endif

// ---- WMMA fragment loaders (16-bit layouts per CDNA5 ISA 7.12.2) ----
// A 16x32: lane m=L%16; halves 0..7 -> K = 8*hi + j ; halves 8..15 -> 16+8*hi+(j-8)
__device__ __forceinline__ v16h frag_a(const half_t* row, int hi) {
  v16h r;
  ((v8h*)&r)[0] = *(const v8h*)(row + 8*hi);
  ((v8h*)&r)[1] = *(const v8h*)(row + 16 + 8*hi);
  return r;
}
// B 32x16: lane n=L%16; halves 0..7 -> K = 8*hi + j ; halves 8..15 -> 16+8*hi+(j-8)
__device__ __forceinline__ v16h frag_b(const half_t* row, int hi) {
  v16h r;
  ((v8h*)&r)[0] = *(const v8h*)(row + 8*hi);
  ((v8h*)&r)[1] = *(const v8h*)(row + 16 + 8*hi);
  return r;
}
__device__ __forceinline__ v8f wmma16(v16h a, v16h b, v8f c) {
  return __builtin_amdgcn_wmma_f32_16x16x32_f16(false, a, false, b, (short)0, c,
                                                false, false);
}

// ---------------------------------------------------------------------------
// Shared double-buffered 128x128x(K=1024) WMMA main loop.
// A: row-major [16384][1024]; Bt: pre-TRANSPOSED weights, row-major [n][k].
// Both tiles are 128 rows x 32 halves -> identical staging.
// ---------------------------------------------------------------------------
__device__ __forceinline__ void gemm_mainloop(const half_t* __restrict__ Abase,
                                              const half_t* __restrict__ Bbase,
                                              half_t* sA, half_t* sB, int tid,
                                              v8f (&acc)[2][4]) {
  const int lane = tid & 31, w = tid >> 5;
  const int ln16 = lane & 15, hi = (lane >> 4) & 1;
  const int wm = w >> 1, wn = w & 1;
  const int NKT = D_MODEL / BK;
#if HAVE_TDM
  if (w == 0) {
    tdm_load_tile_128x32(Abase, lds_addr_of(sA));
    tdm_load_tile_128x32(Bbase, lds_addr_of(sB));
    __builtin_amdgcn_s_wait_tensorcnt(0);
  }
  __syncthreads();
  for (int kt = 0; kt < NKT; ++kt) {
    const int cur = kt & 1, nxt = cur ^ 1;
    if (w == 0 && kt + 1 < NKT) {   // DMA next tiles into back buffer
      tdm_load_tile_128x32(Abase + (kt + 1) * BK,
                           lds_addr_of(sA + nxt * BM * LDT));
      tdm_load_tile_128x32(Bbase + (kt + 1) * BK,
                           lds_addr_of(sB + nxt * BN * LDT));
    }
    const half_t* pA = sA + cur * BM * LDT;
    const half_t* pB = sB + cur * BN * LDT;
    v16h a[2], b[4];
#pragma unroll
    for (int tm = 0; tm < 2; ++tm)
      a[tm] = frag_a(&pA[(wm * 32 + tm * 16 + ln16) * LDT], hi);
#pragma unroll
    for (int tn = 0; tn < 4; ++tn)
      b[tn] = frag_b(&pB[(wn * 64 + tn * 16 + ln16) * LDT], hi);
#pragma unroll
    for (int tm = 0; tm < 2; ++tm)
#pragma unroll
      for (int tn = 0; tn < 4; ++tn)
        acc[tm][tn] = wmma16(a[tm], b[tn], acc[tm][tn]);
    if (w == 0 && kt + 1 < NKT) __builtin_amdgcn_s_wait_tensorcnt(0);
    __syncthreads();
  }
#else
  const int sm = tid >> 1, hseg = (tid & 1) * 16;
  v8h a0, a1, b0, b1;
  {  // prime buffer 0
    const half_t* ga = Abase + (size_t)sm * D_MODEL + hseg;
    const half_t* gb = Bbase + (size_t)sm * D_MODEL + hseg;
    a0 = ((const v8h*)ga)[0]; a1 = ((const v8h*)ga)[1];
    b0 = ((const v8h*)gb)[0]; b1 = ((const v8h*)gb)[1];
  }
  *(v8h*)&sA[sm * LDT + hseg] = a0; *(v8h*)&sA[sm * LDT + hseg + 8] = a1;
  *(v8h*)&sB[sm * LDT + hseg] = b0; *(v8h*)&sB[sm * LDT + hseg + 8] = b1;
  __syncthreads();
  for (int kt = 0; kt < NKT; ++kt) {
    const int cur = kt & 1, nxt = cur ^ 1;
    if (kt + 1 < NKT) {  // issue next-tile global loads before compute
      const half_t* ga = Abase + (size_t)sm * D_MODEL + (kt + 1) * BK + hseg;
      const half_t* gb = Bbase + (size_t)sm * D_MODEL + (kt + 1) * BK + hseg;
      a0 = ((const v8h*)ga)[0]; a1 = ((const v8h*)ga)[1];
      b0 = ((const v8h*)gb)[0]; b1 = ((const v8h*)gb)[1];
      if (kt + 2 < NKT) __builtin_prefetch(ga + BK, 0, 3);
    }
    const half_t* pA = sA + cur * BM * LDT;
    const half_t* pB = sB + cur * BN * LDT;
    v16h a[2], b[4];
#pragma unroll
    for (int tm = 0; tm < 2; ++tm)
      a[tm] = frag_a(&pA[(wm * 32 + tm * 16 + ln16) * LDT], hi);
#pragma unroll
    for (int tn = 0; tn < 4; ++tn)
      b[tn] = frag_b(&pB[(wn * 64 + tn * 16 + ln16) * LDT], hi);
#pragma unroll
    for (int tm = 0; tm < 2; ++tm)
#pragma unroll
      for (int tn = 0; tn < 4; ++tn)
        acc[tm][tn] = wmma16(a[tm], b[tn], acc[tm][tn]);
    if (kt + 1 < NKT) {
      half_t* dA = sA + nxt * BM * LDT;
      half_t* dB = sB + nxt * BN * LDT;
      *(v8h*)&dA[sm * LDT + hseg] = a0; *(v8h*)&dA[sm * LDT + hseg + 8] = a1;
      *(v8h*)&dB[sm * LDT + hseg] = b0; *(v8h*)&dB[sm * LDT + hseg + 8] = b1;
    }
    __syncthreads();
  }
#endif
}

// ---------------------------------------------------------------------------
// Kernel 0a: x fp32 -> f16 (vectorized, exact-size grid)
// ---------------------------------------------------------------------------
__global__ __launch_bounds__(256) void convert_x_kernel(
    const float* __restrict__ x, half_t* __restrict__ xh) {
  size_t n = ((size_t)blockIdx.x * 256 + threadIdx.x) * 4;
  v4f s = *(const v4f*)(x + n);
  v4h d; d[0]=(half_t)s[0]; d[1]=(half_t)s[1]; d[2]=(half_t)s[2]; d[3]=(half_t)s[3];
  *(v4h*)(xh + n) = d;
}

// ---------------------------------------------------------------------------
// Kernel 0b: weights fp32 -> f16 TRANSPOSED (Wt[n][k] = W[k][n]), tiled in LDS
// ---------------------------------------------------------------------------
__global__ __launch_bounds__(256) void convert_wt_kernel(
    const float* __restrict__ Wq, const float* __restrict__ Wk,
    const float* __restrict__ Wv, const float* __restrict__ Wo,
    half_t* __restrict__ Wt) {
  __shared__ float tile[32][33];
  const int wi = blockIdx.z;
  const float* src = (wi == 0) ? Wq : (wi == 1) ? Wk : (wi == 2) ? Wv : Wo;
  const int k0 = blockIdx.y * 32;
  const int n0 = blockIdx.x * 32;
  const int t = threadIdx.x;
  const int r = t >> 3, c4 = (t & 7) * 4;
  v4f s = *(const v4f*)(src + (size_t)(k0 + r) * D_MODEL + n0 + c4);
  tile[r][c4 + 0] = s[0]; tile[r][c4 + 1] = s[1];
  tile[r][c4 + 2] = s[2]; tile[r][c4 + 3] = s[3];
  __syncthreads();
  v4h d;
  d[0] = (half_t)tile[c4 + 0][r]; d[1] = (half_t)tile[c4 + 1][r];
  d[2] = (half_t)tile[c4 + 2][r]; d[3] = (half_t)tile[c4 + 3][r];
  *(v4h*)(Wt + (size_t)wi * D_MODEL * D_MODEL + (size_t)(n0 + r) * D_MODEL +
          k0 + c4) = d;
}

// ---------------------------------------------------------------------------
// Kernel 1: Q/K/V projections (blockIdx.z picks W); fused RoPE+ReLU epilogue.
// ---------------------------------------------------------------------------
__global__ __launch_bounds__(256) void qkv_gemm_kernel(
    const half_t* __restrict__ xh, const half_t* __restrict__ Wt,
    half_t* __restrict__ qh, half_t* __restrict__ kh, half_t* __restrict__ vh) {
  __shared__ __align__(16) half_t sA[2 * BM * LDT];
  __shared__ __align__(16) half_t sB[2 * BN * LDT];
  const int tid = threadIdx.x;
  const int lane = tid & 31, w = tid >> 5;
  const int ln16 = lane & 15, hi = (lane >> 4) & 1;
  const int wm = w >> 1, wn = w & 1;
  const int mode = blockIdx.z;
  const int row0 = blockIdx.y * BM;
  const int n0 = blockIdx.x * BN;
  const half_t* W = Wt + (size_t)mode * D_MODEL * D_MODEL;
  half_t* dst = (mode == 0) ? qh : (mode == 1) ? kh : vh;

  v8f acc[2][4] = {};
  gemm_mainloop(xh + (size_t)row0 * D_MODEL, W + (size_t)n0 * D_MODEL, sA, sB,
                tid, acc);

  // Epilogue: modes 0/1 apply RoPE (pair partner = adjacent lane) + ReLU.
#pragma unroll
  for (int tm = 0; tm < 2; ++tm)
#pragma unroll
    for (int tn = 0; tn < 4; ++tn)
#pragma unroll
      for (int g = 0; g < 8; ++g) {
        float val = acc[tm][tn][g];
        const int mg = row0 + wm * 32 + tm * 16 + g + 8 * hi;
        const int ng = n0 + wn * 64 + tn * 16 + ln16;
        const int b_ = mg >> 12, s = mg & (S_LEN - 1);
        const int h = ng >> 6, d = ng & (HDIM - 1);
        float outv;
        if (mode < 2) {
          float part = __shfl_xor(val, 1, 32);  // even<->odd feature partner
          float ang = (float)s *
                      exp2f(-(float)(d & ~1) * (LOG2_THETA / (float)HDIM));
          float sn, cs;
          __sincosf(ang, &sn, &cs);
          float r = (d & 1) ? (part * sn + val * cs) : (val * cs - part * sn);
          outv = fmaxf(r, 0.0f);
        } else {
          outv = val;
        }
        dst[(((size_t)(b_ * NHEADS + h)) * S_LEN + s) * HDIM + d] =
            (half_t)outv;
      }
}

// ---------------------------------------------------------------------------
// Kernel 2: per (b,h): vk[64][64] = v^T k over S=4096 (WMMA) + ksum = colsum(k)
// (ksum is the reference's "pad column", computed without padding).
// ---------------------------------------------------------------------------
__global__ __launch_bounds__(256) void vk_kernel(
    const half_t* __restrict__ kh, const half_t* __restrict__ vh,
    half_t* __restrict__ vkh, half_t* __restrict__ ksumh) {
  __shared__ __align__(16) half_t sV[64 * LDT];  // transposed [e][s]
  __shared__ __align__(16) half_t sK[64 * LDT];  // transposed [d][s]
  __shared__ float sredk[64 * 32];               // ksum reduction scratch
  const int tid = threadIdx.x;
  const int lane = tid & 31, w = tid >> 5;
  const int ln16 = lane & 15, hi = (lane >> 4) & 1;
  const int bh = blockIdx.x;
  const size_t base = (size_t)bh * S_LEN * HDIM;
  const int te = w >> 1, tdp = (w & 1) * 2;  // wave -> 2 of 16 output tiles
  const int ss = tid >> 3;                   // staging s row 0..31
  const int e0 = (tid & 7) * 8;              // staging col group

  v8f acc[2] = {};
  float part8[8] = {0, 0, 0, 0, 0, 0, 0, 0};

  for (int st = 0; st < S_LEN / 32; ++st) {
    const int s0 = st * 32;
    v8h kv = *(const v8h*)(kh + base + (size_t)(s0 + ss) * HDIM + e0);
    v8h vv = *(const v8h*)(vh + base + (size_t)(s0 + ss) * HDIM + e0);
#pragma unroll
    for (int i = 0; i < 8; ++i) {
      sK[(e0 + i) * LDT + ss] = kv[i];
      sV[(e0 + i) * LDT + ss] = vv[i];
      part8[i] += (float)kv[i];  // ksum partial straight from registers
    }
    __syncthreads();
    v16h a = frag_a(&sV[(te * 16 + ln16) * LDT], hi);
#pragma unroll
    for (int p = 0; p < 2; ++p) {
      v16h b = frag_b(&sK[((tdp + p) * 16 + ln16) * LDT], hi);
      acc[p] = wmma16(a, b, acc[p]);
    }
    __syncthreads();
  }
#pragma unroll
  for (int p = 0; p < 2; ++p)
#pragma unroll
    for (int g = 0; g < 8; ++g) {
      const int e = te * 16 + g + 8 * hi;
      const int d = (tdp + p) * 16 + ln16;
      vkh[(size_t)bh * HDIM * HDIM + e * HDIM + d] = (half_t)acc[p][g];
    }
#pragma unroll
  for (int i = 0; i < 8; ++i) sredk[(e0 + i) * 32 + ss] = part8[i];
  __syncthreads();
  if (tid < 64) {
    float s = 0.0f;
#pragma unroll
    for (int j = 0; j < 32; ++j) s += sredk[tid * 32 + j];
    ksumh[bh * HDIM + tid] = (half_t)s;
  }
}

// ---------------------------------------------------------------------------
// Kernel 3: num = q @ vk^T (WMMA), denom = q . ksum, normalize, merge heads.
// ---------------------------------------------------------------------------
__global__ __launch_bounds__(256) void attn_kernel(
    const half_t* __restrict__ qh, const half_t* __restrict__ vkh,
    const half_t* __restrict__ ksumh, half_t* __restrict__ attn) {
  __shared__ __align__(16) half_t svk[64 * LDVK];
  __shared__ float sksum[64];
  __shared__ float sden[128];
  const int tid = threadIdx.x;
  const int lane = tid & 31, w = tid >> 5;
  const int ln16 = lane & 15, hi = (lane >> 4) & 1;
  const int bh = blockIdx.y;
  const int b_ = bh >> 4, h = bh & 15;
  const int s0 = blockIdx.x * 128;
  const size_t qbase = (size_t)bh * S_LEN * HDIM;

  {  // load vk (row-major [e][d]) into LDS
    const int e = tid >> 2, c0 = (tid & 3) * 16;
    const half_t* g = vkh + (size_t)bh * HDIM * HDIM + e * HDIM + c0;
    *(v8h*)&svk[e * LDVK + c0]     = *(const v8h*)(g);
    *(v8h*)&svk[e * LDVK + c0 + 8] = *(const v8h*)(g + 8);
  }
  if (tid < 64) sksum[tid] = (float)ksumh[bh * HDIM + tid];
  __syncthreads();

  v8f acc[4] = {};
  const int srow = s0 + w * 16 + ln16;
#pragma unroll
  for (int ks = 0; ks < 2; ++ks) {
    v16h a;  // A fragment straight from global q (row-major, contiguous K)
    const half_t* qrow = qh + qbase + (size_t)srow * HDIM + 32 * ks;
    ((v8h*)&a)[0] = *(const v8h*)(qrow + 8 * hi);
    ((v8h*)&a)[1] = *(const v8h*)(qrow + 16 + 8 * hi);
#pragma unroll
    for (int tn = 0; tn < 4; ++tn) {
      v16h b = frag_b(&svk[(tn * 16 + ln16) * LDVK + 32 * ks], hi);
      acc[tn] = wmma16(a, b, acc[tn]);
    }
  }
  if (tid < 128) {  // denominator for each of the 128 rows
    const half_t* qrow = qh + qbase + (size_t)(s0 + tid) * HDIM;
    float sum = 0.0f;
#pragma unroll
    for (int c = 0; c < 8; ++c) {
      v8h qv = *(const v8h*)(qrow + c * 8);
#pragma unroll
      for (int i = 0; i < 8; ++i) sum += (float)qv[i] * sksum[c * 8 + i];
    }
    sden[tid] = sum;
  }
  __syncthreads();
#pragma unroll
  for (int tn = 0; tn < 4; ++tn)
#pragma unroll
    for (int g = 0; g < 8; ++g) {
      const int mloc = w * 16 + g + 8 * hi;
      const float val = acc[tn][g] / (sden[mloc] + 1e-6f);
      const int e = tn * 16 + ln16;
      attn[((size_t)(b_ * S_LEN + s0 + mloc)) * D_MODEL + h * HDIM + e] =
          (half_t)val;
    }
}

// ---------------------------------------------------------------------------
// Kernel 4: out = attn @ Wo  (fp32 result)
// ---------------------------------------------------------------------------
__global__ __launch_bounds__(256) void out_gemm_kernel(
    const half_t* __restrict__ ah, const half_t* __restrict__ Wot,
    float* __restrict__ out) {
  __shared__ __align__(16) half_t sA[2 * BM * LDT];
  __shared__ __align__(16) half_t sB[2 * BN * LDT];
  const int tid = threadIdx.x;
  const int lane = tid & 31, w = tid >> 5;
  const int ln16 = lane & 15, hi = (lane >> 4) & 1;
  const int wm = w >> 1, wn = w & 1;
  const int row0 = blockIdx.y * BM;
  const int n0 = blockIdx.x * BN;

  v8f acc[2][4] = {};
  gemm_mainloop(ah + (size_t)row0 * D_MODEL, Wot + (size_t)n0 * D_MODEL, sA, sB,
                tid, acc);
#pragma unroll
  for (int tm = 0; tm < 2; ++tm)
#pragma unroll
    for (int tn = 0; tn < 4; ++tn)
#pragma unroll
      for (int g = 0; g < 8; ++g) {
        const int mg = row0 + wm * 32 + tm * 16 + g + 8 * hi;
        const int ng = n0 + wn * 64 + tn * 16 + ln16;
        out[(size_t)mg * D_MODEL + ng] = acc[tm][tn][g];
      }
}

// ---------------------------------------------------------------------------
extern "C" void kernel_launch(void* const* d_in, const int* in_sizes, int n_in,
                              void* d_out, int out_size, void* d_ws,
                              size_t ws_size, hipStream_t stream) {
  const float* x  = (const float*)d_in[0];
  const float* Wq = (const float*)d_in[1];
  const float* Wk = (const float*)d_in[2];
  const float* Wv = (const float*)d_in[3];
  const float* Wo = (const float*)d_in[4];
  float* out = (float*)d_out;

  const size_t NX = (size_t)NROWS * D_MODEL;      // 16,777,216
  const size_t NW = (size_t)D_MODEL * D_MODEL;    // 1,048,576
  char* ws = (char*)d_ws;
  half_t* xh = (half_t*)ws;                       // 32 MB; reused as attn buf
  half_t* Wt = (half_t*)(ws + NX * 2);            // 8 MB transposed f16 weights
  half_t* qh = (half_t*)(ws + NX * 2 + 4 * NW * 2);
  half_t* kh = qh + NX;
  half_t* vh = kh + NX;
  half_t* vkh = vh + NX;                          // 64 * 64 * 64 f16
  half_t* ksumh = vkh + (size_t)BATCH * NHEADS * HDIM * HDIM;

  convert_x_kernel<<<dim3((unsigned)(NX / 4 / 256)), 256, 0, stream>>>(x, xh);
  convert_wt_kernel<<<dim3(D_MODEL / 32, D_MODEL / 32, 4), 256, 0, stream>>>(
      Wq, Wk, Wv, Wo, Wt);
  qkv_gemm_kernel<<<dim3(D_MODEL / BN, NROWS / BM, 3), 256, 0, stream>>>(
      xh, Wt, qh, kh, vh);
  vk_kernel<<<dim3(BATCH * NHEADS), 256, 0, stream>>>(kh, vh, vkh, ksumh);
  attn_kernel<<<dim3(S_LEN / 128, BATCH * NHEADS), 256, 0, stream>>>(
      qh, vkh, ksumh, xh /* attn reuses xh */);
  out_gemm_kernel<<<dim3(D_MODEL / BN, NROWS / BM), 256, 0, stream>>>(
      xh, Wt + 3 * NW, out);
}
